// ConstantTimeStrideAttention_70635032150394
// MI455X (gfx1250) — compile-verified, matrix operand order checked by reading, not run on hardware
//
#include <hip/hip_runtime.h>

// ---------------------------------------------------------------------------
// Problem constants (reference: B=2, S=2048, DIM=1536, H=12, HD=128, SIGMA=12)
// ---------------------------------------------------------------------------
#define B_    2
#define S_    2048
#define DIM_  1536
#define H_    12
#define HD_   128
#define M_    (B_ * S_)     // 4096 rows of activations
#define N1_   (3 * DIM_)    // 4608 qkv columns

typedef __attribute__((ext_vector_type(16))) _Float16       v16h;
typedef __attribute__((ext_vector_type(8)))  float          v8f;
typedef __attribute__((ext_vector_type(8)))  unsigned short v8u16;

union Frag16 { v8u16 h[2]; v16h v; };
struct H4 { _Float16 x, y, z, w; };

// ---------------------------------------------------------------------------
// Pack fp32 activations -> f16 (vectorized float4 -> 4x f16)
// ---------------------------------------------------------------------------
__global__ void pack_x_f16(const float* __restrict__ in, _Float16* __restrict__ out, int n4) {
    int i = blockIdx.x * blockDim.x + threadIdx.x;
    if (i >= n4) return;
    float4 f = ((const float4*)in)[i];
    H4 o;
    o.x = (_Float16)f.x; o.y = (_Float16)f.y; o.z = (_Float16)f.z; o.w = (_Float16)f.w;
    ((H4*)out)[i] = o;
}

// ---------------------------------------------------------------------------
// LDS-tiled transpose + pack: in is (K x N) fp32 row-major, out is (N x K) f16
// so that WMMA B-fragments become contiguous 16B loads per lane.
// ---------------------------------------------------------------------------
__global__ void transpose_pack_f16(const float* __restrict__ in, _Float16* __restrict__ out,
                                   int K, int N) {
    __shared__ float tile[32][33];   // +1 pad kills LDS bank conflicts
    int x  = blockIdx.x * 32 + threadIdx.x;   // column n (input)
    int y0 = blockIdx.y * 32 + threadIdx.y;   // row k base (input)
#pragma unroll
    for (int j = 0; j < 32; j += 8) {
        int y = y0 + j;
        if (x < N && y < K) tile[threadIdx.y + j][threadIdx.x] = in[(size_t)y * N + x];
    }
    __syncthreads();
    int xo  = blockIdx.y * 32 + threadIdx.x;  // k (output col)
    int yo0 = blockIdx.x * 32 + threadIdx.y;  // n (output row)
#pragma unroll
    for (int j = 0; j < 32; j += 8) {
        int yo = yo0 + j;
        if (xo < K && yo < N)
            out[(size_t)yo * K + xo] = (_Float16)tile[threadIdx.x][threadIdx.y + j];
    }
}

// ---------------------------------------------------------------------------
// WMMA GEMM:  C(MxN) = A(MxK, f16) * Bt(NxK, f16)^T + bias(N),  f32 accumulate.
//
// Block = 256 threads = 8 waves arranged 4(M) x 2(N); each wave owns a 32x64
// strip of C (2 A-fragments x 4 B-fragments = 8 WMMAs per 32-wide K step).
// The K loop is software-pipelined with two register fragment buffers and a
// step of 64: loads for chunk n+1 are issued before the WMMAs for chunk n so
// the matrix pipes overlap L2 latency instead of stalling on loadcnt==0.
//
// Requires M%128==0, N%128==0, K%64==0 (true: 4096 / 4608|1536 / 1536).
//
// Lane layout for 16-bit fragments (ISA 7.12.2): lane r=lane&15 holds row
// m0+r; lanes 0-15 hold K {0..7,16..23}, lanes 16-31 hold K {8..15,24..31}
// => two contiguous v8u16 (16B) loads per fragment. B mirrors A over the
// N-major (transposed) weight storage.
// ---------------------------------------------------------------------------
__launch_bounds__(256)
__global__ void gemm_wmma_f16(const _Float16* __restrict__ A,
                              const _Float16* __restrict__ Bt,
                              const float* __restrict__ bias,
                              float* __restrict__ C,
                              int M, int N, int K) {
    const int lane = threadIdx.x & 31;
    const int wave = threadIdx.x >> 5;
    const int mw   = wave & 3;                // 4 waves along M
    const int nw   = wave >> 2;               // 2 waves along N
    const int r    = lane & 15;
    const int hi   = lane >> 4;               // 0: K {0..7,16..23}, 1: K {8..15,24..31}
    const int n0   = blockIdx.x * 128 + nw * 64;
    const int m0   = blockIdx.y * 128 + mw * 32;

    const v8u16* ap0 = (const v8u16*)(A  + (size_t)(m0 +      r) * K);
    const v8u16* ap1 = (const v8u16*)(A  + (size_t)(m0 + 16 + r) * K);
    const v8u16* bp0 = (const v8u16*)(Bt + (size_t)(n0 +      r) * K);
    const v8u16* bp1 = (const v8u16*)(Bt + (size_t)(n0 + 16 + r) * K);
    const v8u16* bp2 = (const v8u16*)(Bt + (size_t)(n0 + 32 + r) * K);
    const v8u16* bp3 = (const v8u16*)(Bt + (size_t)(n0 + 48 + r) * K);

    v8f acc00 = {}, acc01 = {}, acc02 = {}, acc03 = {};
    v8f acc10 = {}, acc11 = {}, acc12 = {}, acc13 = {};

    Frag16 a0A, a1A, b0A, b1A, b2A, b3A;   // buffer A
    Frag16 a0B, a1B, b0B, b1B, b2B, b3B;   // buffer B

    auto load_set = [&](int k0, Frag16& a0, Frag16& a1,
                        Frag16& b0, Frag16& b1, Frag16& b2, Frag16& b3) {
        const int i0 = (k0 >> 3) + hi;     // 16B chunk: K k0+hi*8 .. +7
        const int i1 = i0 + 2;             // 16B chunk: K k0+16+hi*8 ..
        a0.h[0] = ap0[i0]; a0.h[1] = ap0[i1];
        a1.h[0] = ap1[i0]; a1.h[1] = ap1[i1];
        b0.h[0] = bp0[i0]; b0.h[1] = bp0[i1];
        b1.h[0] = bp1[i0]; b1.h[1] = bp1[i1];
        b2.h[0] = bp2[i0]; b2.h[1] = bp2[i1];
        b3.h[0] = bp3[i0]; b3.h[1] = bp3[i1];
    };
    auto do_wmma = [&](Frag16& a0, Frag16& a1,
                       Frag16& b0, Frag16& b1, Frag16& b2, Frag16& b3) {
        acc00 = __builtin_amdgcn_wmma_f32_16x16x32_f16(false, a0.v, false, b0.v,
                                                       (short)0, acc00, false, false);
        acc01 = __builtin_amdgcn_wmma_f32_16x16x32_f16(false, a0.v, false, b1.v,
                                                       (short)0, acc01, false, false);
        acc02 = __builtin_amdgcn_wmma_f32_16x16x32_f16(false, a0.v, false, b2.v,
                                                       (short)0, acc02, false, false);
        acc03 = __builtin_amdgcn_wmma_f32_16x16x32_f16(false, a0.v, false, b3.v,
                                                       (short)0, acc03, false, false);
        acc10 = __builtin_amdgcn_wmma_f32_16x16x32_f16(false, a1.v, false, b0.v,
                                                       (short)0, acc10, false, false);
        acc11 = __builtin_amdgcn_wmma_f32_16x16x32_f16(false, a1.v, false, b1.v,
                                                       (short)0, acc11, false, false);
        acc12 = __builtin_amdgcn_wmma_f32_16x16x32_f16(false, a1.v, false, b2.v,
                                                       (short)0, acc12, false, false);
        acc13 = __builtin_amdgcn_wmma_f32_16x16x32_f16(false, a1.v, false, b3.v,
                                                       (short)0, acc13, false, false);
    };

    // software pipeline, distance 1, two buffers, step 64 (K/32 is even)
    load_set(0, a0A, a1A, b0A, b1A, b2A, b3A);
    int k0 = 0;
    for (; k0 + 64 < K; k0 += 64) {
        load_set(k0 + 32, a0B, a1B, b0B, b1B, b2B, b3B);
        do_wmma(a0A, a1A, b0A, b1A, b2A, b3A);
        load_set(k0 + 64, a0A, a1A, b0A, b1A, b2A, b3A);
        do_wmma(a0B, a1B, b0B, b1B, b2B, b3B);
    }
    load_set(k0 + 32, a0B, a1B, b0B, b1B, b2B, b3B);
    do_wmma(a0A, a1A, b0A, b1A, b2A, b3A);
    do_wmma(a0B, a1B, b0B, b1B, b2B, b3B);

    // Epilogue: C/D layout (ISA 7.12.2): VGPR v holds row v + 8*hi, lane%16 = column
    const float bv0 = bias[n0 +      r];
    const float bv1 = bias[n0 + 16 + r];
    const float bv2 = bias[n0 + 32 + r];
    const float bv3 = bias[n0 + 48 + r];
    float* crow0 = C + (size_t)(m0 +      hi * 8) * N;
    float* crow1 = C + (size_t)(m0 + 16 + hi * 8) * N;
#pragma unroll
    for (int v = 0; v < 8; ++v) {
        float* cp0 = crow0 + (size_t)v * N;
        float* cp1 = crow1 + (size_t)v * N;
        cp0[n0 +      r] = acc00[v] + bv0;
        cp0[n0 + 16 + r] = acc01[v] + bv1;
        cp0[n0 + 32 + r] = acc02[v] + bv2;
        cp0[n0 + 48 + r] = acc03[v] + bv3;
        cp1[n0 +      r] = acc10[v] + bv0;
        cp1[n0 + 16 + r] = acc11[v] + bv1;
        cp1[n0 + 32 + r] = acc12[v] + bv2;
        cp1[n0 + 48 + r] = acc13[v] + bv3;
    }
}

// ---------------------------------------------------------------------------
// Anchor attention: one wave32 per (b,h,s). Lane owns 4 of the 128 head dims.
// qkv is (M x 4608) fp32: cols [0,1536)=Q, [1536,3072)=K, [3072,4608)=V,
// within each group col = h*128 + d. Output written as f16 (M x 1536),
// col = h*128 + d  (matches transpose(0,2,1,3).reshape in reference).
// ---------------------------------------------------------------------------
__launch_bounds__(256)
__global__ void anchor_attn(const float* __restrict__ qkv,
                            const float* __restrict__ gscale,
                            _Float16* __restrict__ attn_out) {
    const int lane = threadIdx.x & 31;
    const int wid  = blockIdx.x * 8 + (threadIdx.x >> 5);
    const int s = wid & (S_ - 1);
    const int h = (wid >> 11) % H_;
    const int b = wid / (S_ * H_);

    // anchor indices: 6 local, 4 strided (step 5), 2 global
    const int offs[10] = {-3, -2, -1, 1, 2, 3, -10, -5, 5, 10};
    int idxs[12];
#pragma unroll
    for (int a = 0; a < 10; ++a) {
        int t = s + offs[a];
        idxs[a] = min(max(t, 0), S_ - 1);
    }
    idxs[10] = 0;
    idxs[11] = S_ - 1;

    const int coff = h * HD_ + lane * 4;
    const size_t rowq = (size_t)(b * S_ + s) * N1_;
    const float4 q = *(const float4*)(qkv + rowq + coff);

    float sc[12];
#pragma unroll
    for (int a = 0; a < 12; ++a) {
        const size_t rk = (size_t)(b * S_ + idxs[a]) * N1_ + DIM_ + coff;
        const float4 kv = *(const float4*)(qkv + rk);
        float p = q.x * kv.x + q.y * kv.y + q.z * kv.z + q.w * kv.w;
#pragma unroll
        for (int o = 16; o > 0; o >>= 1) p += __shfl_xor(p, o, 32);  // wave32 butterfly
        sc[a] = p;
    }

    // group-weight log bias: softmax over the 3 group scales
    const float g0 = gscale[0], g1 = gscale[1], g2 = gscale[2];
    const float mg = fmaxf(g0, fmaxf(g1, g2));
    const float e0 = __expf(g0 - mg), e1 = __expf(g1 - mg), e2 = __expf(g2 - mg);
    const float ls = __logf(e0 + e1 + e2);
    const float lw0 = g0 - mg - ls, lw1 = g1 - mg - ls, lw2 = g2 - mg - ls;
    const float scale = 0.088388347648318447f;  // 128^-0.5

#pragma unroll
    for (int a = 0; a < 12; ++a)
        sc[a] = sc[a] * scale + (a < 6 ? lw0 : (a < 10 ? lw1 : lw2));

    float mx = sc[0];
#pragma unroll
    for (int a = 1; a < 12; ++a) mx = fmaxf(mx, sc[a]);
    float sum = 0.f;
#pragma unroll
    for (int a = 0; a < 12; ++a) { sc[a] = __expf(sc[a] - mx); sum += sc[a]; }
    const float inv = 1.0f / sum;

    float4 o = {0.f, 0.f, 0.f, 0.f};
#pragma unroll
    for (int a = 0; a < 12; ++a) {
        const size_t rv = (size_t)(b * S_ + idxs[a]) * N1_ + 2 * DIM_ + coff;
        const float4 vv = *(const float4*)(qkv + rv);
        const float p = sc[a] * inv;
        o.x += p * vv.x; o.y += p * vv.y; o.z += p * vv.z; o.w += p * vv.w;
    }

    H4 hv;
    hv.x = (_Float16)o.x; hv.y = (_Float16)o.y; hv.z = (_Float16)o.z; hv.w = (_Float16)o.w;
    *(H4*)(attn_out + (size_t)(b * S_ + s) * DIM_ + coff) = hv;
}

// ---------------------------------------------------------------------------
// Launcher
// ---------------------------------------------------------------------------
extern "C" void kernel_launch(void* const* d_in, const int* in_sizes, int n_in,
                              void* d_out, int out_size, void* d_ws, size_t ws_size,
                              hipStream_t stream) {
    const float* x      = (const float*)d_in[0];  // (B,S,DIM)
    const float* Wqkv   = (const float*)d_in[1];  // (DIM, 3*DIM)
    const float* bqkv   = (const float*)d_in[2];  // (3*DIM)
    const float* Wout   = (const float*)d_in[3];  // (DIM, DIM)
    const float* bout   = (const float*)d_in[4];  // (DIM)
    const float* gscale = (const float*)d_in[5];  // (3)
    float* out = (float*)d_out;

    char* ws = (char*)d_ws;
    size_t off = 0;
    auto take = [&](size_t bytes) { size_t p = off; off += (bytes + 255) & ~(size_t)255; return p; };
    _Float16* Xh   = (_Float16*)(ws + take((size_t)M_ * DIM_ * 2));   // 12.6 MB
    _Float16* Wqh  = (_Float16*)(ws + take((size_t)N1_ * DIM_ * 2));  // 14.2 MB
    _Float16* Woh  = (_Float16*)(ws + take((size_t)DIM_ * DIM_ * 2)); //  4.7 MB
    float*    qkvb = (float*)   (ws + take((size_t)M_ * N1_ * 4));    // 75.5 MB
    _Float16* Ah   = Xh;  // reuse: X-f16 is dead after GEMM1

    // 1) pack activations to f16
    pack_x_f16<<<(M_ * DIM_ / 4 + 255) / 256, 256, 0, stream>>>(x, Xh, M_ * DIM_ / 4);

    // 2) transpose+pack weights to N-major f16
    dim3 tb(32, 8);
    transpose_pack_f16<<<dim3(N1_ / 32, DIM_ / 32), tb, 0, stream>>>(Wqkv, Wqh, DIM_, N1_);
    transpose_pack_f16<<<dim3(DIM_ / 32, DIM_ / 32), tb, 0, stream>>>(Wout, Woh, DIM_, DIM_);

    // 3) QKV projection: (4096x1536) @ (1536x4608) + bqkv -> fp32 qkv
    gemm_wmma_f16<<<dim3(N1_ / 128, M_ / 128), 256, 0, stream>>>(Xh, Wqh, bqkv, qkvb,
                                                                 M_, N1_, DIM_);

    // 4) anchor attention -> f16 (4096x1536)
    anchor_attn<<<(B_ * H_ * S_) / 8, 256, 0, stream>>>(qkvb, gscale, Ah);

    // 5) output projection: (4096x1536) @ (1536x1536) + bout -> fp32 out
    gemm_wmma_f16<<<dim3(DIM_ / 128, M_ / 128), 256, 0, stream>>>(Ah, Woh, bout, out,
                                                                  M_, DIM_, DIM_);
}
